// BaseMoeModule_83081847374407
// MI455X (gfx1250) — compile-verified, matrix operand order checked by reading, not run on hardware
//
#include <hip/hip_runtime.h>
#include <hip/hip_bf16.h>

// ---------------- problem constants ----------------
#define T_TOK 2048
#define H_DIM 768
#define E_EXP 8
#define K_TOP 2
#define F_DIM 3072
#define TK (T_TOK * K_TOP)          // 4096 assignments
#define MB 32                       // rows (assignments) per block
#define MT_PER_E (TK / MB)          // 128 m-tiles per expert (worst case)
#define F_CHUNK 384                 // F processed in 8 chunks
#define N_CHUNKS (F_DIM / F_CHUNK)  // 8

// ---------------- WMMA types ----------------
typedef __attribute__((ext_vector_type(16))) __bf16 v16bf;
typedef __attribute__((ext_vector_type(8)))  float  v8f;

union FragU { uint4 u[2]; v16bf v; };

__device__ __forceinline__ unsigned short f2bf(float f) {
  unsigned int u = __float_as_uint(f);
  unsigned int r = u + 0x7FFFu + ((u >> 16) & 1u);   // RNE
  return (unsigned short)(r >> 16);
}

// 16-element bf16 WMMA fragment: per ISA 16-bit A/B layout, lanes 0-15 hold
// K=[0..7]+[16..23], lanes 16-31 hold K=[8..15]+[24..31]; caller folds
// kb = (lane>=16)?8:0 into p.
__device__ __forceinline__ v16bf load_frag(const unsigned short* p) {
  FragU f;
  f.u[0] = *(const uint4*)(p);        // K +0..+7
  f.u[1] = *(const uint4*)(p + 16);   // K +16..+23
  return f.v;
}

// ---------------- small init ----------------
__global__ void k_init(int* counts) {
  if (threadIdx.x < E_EXP) counts[threadIdx.x] = 0;
}

// ---------------- router: logits, softmax-top2, bucket build ----------------
__global__ void k_router(const float* __restrict__ x,
                         const float* __restrict__ rw,
                         float* __restrict__ assign_w,
                         int* __restrict__ counts,
                         int* __restrict__ bucket) {
  const int lane = threadIdx.x & 31;
  const int wave = threadIdx.x >> 5;
  const int t = blockIdx.x * 8 + wave;

  float acc[E_EXP];
#pragma unroll
  for (int e = 0; e < E_EXP; ++e) acc[e] = 0.f;

  for (int h = lane; h < H_DIM; h += 32) {
    const float xv = x[(size_t)t * H_DIM + h];
    const float4* r4 = (const float4*)(rw + (size_t)h * E_EXP);
    float4 a = r4[0], b = r4[1];
    acc[0] += xv * a.x; acc[1] += xv * a.y; acc[2] += xv * a.z; acc[3] += xv * a.w;
    acc[4] += xv * b.x; acc[5] += xv * b.y; acc[6] += xv * b.z; acc[7] += xv * b.w;
  }
#pragma unroll
  for (int off = 16; off >= 1; off >>= 1) {
#pragma unroll
    for (int e = 0; e < E_EXP; ++e) acc[e] += __shfl_xor(acc[e], off, 32);
  }

  if (lane == 0) {
    float l0 = -1e30f, l1 = -1e30f; int i0 = 0, i1 = 0;
#pragma unroll
    for (int e = 0; e < E_EXP; ++e) {
      float v = acc[e];
      if (v > l0) { l1 = l0; i1 = i0; l0 = v; i0 = e; }
      else if (v > l1) { l1 = v; i1 = e; }
    }
    // normalized top-2 softmax weights: w0 = exp(l0)/(exp(l0)+exp(l1))
    const float w0 = 1.f / (1.f + __expf(l1 - l0));
    assign_w[t * 2 + 0] = w0;
    assign_w[t * 2 + 1] = 1.f - w0;
    int p0 = atomicAdd(&counts[i0], 1);
    bucket[i0 * TK + p0] = t * 2 + 0;
    int p1 = atomicAdd(&counts[i1], 1);
    bucket[i1 * TK + p1] = t * 2 + 1;
  }
}

// ---------------- x fp32 -> bf16 ----------------
__global__ void k_convert_x(const float* __restrict__ x,
                            unsigned short* __restrict__ xb) {
  const size_t i4 = (size_t)blockIdx.x * blockDim.x + threadIdx.x;  // float4 index
  float4 v = ((const float4*)x)[i4];
  uint2 o;
  o.x = (unsigned int)f2bf(v.x) | ((unsigned int)f2bf(v.y) << 16);
  o.y = (unsigned int)f2bf(v.z) | ((unsigned int)f2bf(v.w) << 16);
  ((uint2*)xb)[i4] = o;
}

// ---------------- per-expert [R][C] fp32 -> [C][R] bf16 (tiled transpose) ----
__global__ void k_transpose_bf16(const float* __restrict__ in,
                                 unsigned short* __restrict__ out,
                                 int R, int C) {
  __shared__ float s[32][33];
  const int tilesC = C / 32;
  const int per = (R / 32) * tilesC;
  const int e  = blockIdx.x / per;
  const int r2 = blockIdx.x % per;
  const int rt = r2 / tilesC;
  const int ct = r2 % tilesC;
  const float* inE = in + (size_t)e * R * C;
  unsigned short* outE = out + (size_t)e * R * C;

  const int r0 = threadIdx.x >> 5;   // 0..7
  const int c0 = threadIdx.x & 31;   // 0..31
#pragma unroll
  for (int i = 0; i < 4; ++i) {
    int r = r0 + 8 * i;
    s[r][c0] = inE[(size_t)(rt * 32 + r) * C + ct * 32 + c0];
  }
  __syncthreads();
#pragma unroll
  for (int i = 0; i < 4; ++i) {
    int cc = r0 + 8 * i;             // column of input = row of output
    outE[(size_t)(ct * 32 + cc) * R + rt * 32 + c0] = f2bf(s[c0][cc]);
  }
}

// ---------------- fused expert MLP: y = silu(x@W1) @ W2, tensor cores -------
#define XS 776   // x LDS row stride (768 + 8 bf16 pad)
#define HS 392   // h LDS row stride (384 + 8 bf16 pad)

__global__ __launch_bounds__(256)
void k_moe_mlp(const unsigned short* __restrict__ xb,
               const unsigned short* __restrict__ w1T,   // [E][F][H] bf16
               const unsigned short* __restrict__ w2T,   // [E][H][F] bf16
               const float* __restrict__ assign_w,
               const int* __restrict__ counts,
               const int* __restrict__ bucket,
               float* __restrict__ ys) {
  __shared__ unsigned short sX[MB * XS];   // x tile   32 x 768 bf16
  __shared__ unsigned short sH[MB * HS];   // h chunk  32 x 384 bf16
  __shared__ int   sA[MB];
  __shared__ float sW[MB];

  const int e  = blockIdx.x >> 7;          // 1024 blocks: 8 experts x 128 tiles
  const int mt = blockIdx.x & 127;
  const int cnt = counts[e];
  if (mt * MB >= cnt) return;

  const int tid  = threadIdx.x;
  const int lane = tid & 31;
  const int wave = tid >> 5;
  const int mloc = lane & 15;          // A-row / B-row / C-col within tile
  const int kb   = (lane >> 4) * 8;    // K sub-offset per lane half
  const int mrow = (lane >> 4) << 3;   // C-layout M base

  if (tid < MB) {
    int idx = mt * MB + tid; if (idx > cnt - 1) idx = cnt - 1;
    int a = bucket[e * TK + idx];
    sA[tid] = a;
    sW[tid] = assign_w[a];
  }
  __syncthreads();

  // cooperative gather of 32 x rows (bf16) into LDS, 16B chunks
  for (int c = tid; c < MB * 96; c += 256) {
    int m = c / 96, q = c % 96;
    int t = sA[m] >> 1;
    *(uint4*)(&sX[m * XS + q * 8]) =
        *(const uint4*)(xb + (size_t)t * H_DIM + q * 8);
  }
  __syncthreads();

  // phase-B accumulators: 6 n-tiles x 2 m-subtiles, persistent over chunks
  v8f acc[12];
#pragma unroll
  for (int j = 0; j < 12; ++j) {
#pragma unroll
    for (int v = 0; v < 8; ++v) acc[j][v] = 0.f;
  }

  const unsigned short* aX0 = &sX[mloc * XS];          // rows 0..15
  const unsigned short* aX1 = &sX[(mloc + 16) * XS];   // rows 16..31
  const unsigned short* aH0 = &sH[mloc * HS];
  const unsigned short* aH1 = &sH[(mloc + 16) * HS];

  for (int chunk = 0; chunk < N_CHUNKS; ++chunk) {
    // ---- phase A: h[:, chunk] = silu(x @ W1[:, chunk]) ----
    // wave owns n-tiles (wave*3 + j), j=0..2, within this 384-wide chunk
    v8f ha[6];
#pragma unroll
    for (int j = 0; j < 6; ++j) {
#pragma unroll
      for (int v = 0; v < 8; ++v) ha[j][v] = 0.f;
    }
    const unsigned short* b1p[3];
#pragma unroll
    for (int j = 0; j < 3; ++j) {
      const int nf = chunk * F_CHUNK + (wave * 3 + j) * 16 + mloc;  // F index
      b1p[j] = w1T + ((size_t)e * F_DIM + nf) * H_DIM + kb;
    }
    for (int ks = 0; ks < 24; ++ks) {          // K = H = 768
      const int k0 = ks * 32;
      v16bf a0 = load_frag(aX0 + k0 + kb);
      v16bf a1 = load_frag(aX1 + k0 + kb);
#pragma unroll
      for (int j = 0; j < 3; ++j) {
        __builtin_prefetch(b1p[j] + k0 + 128, 0, 3);
        v16bf b = load_frag(b1p[j] + k0);
        ha[j * 2 + 0] = __builtin_amdgcn_wmma_f32_16x16x32_bf16(
            false, a0, false, b, (short)0, ha[j * 2 + 0], false, false);
        ha[j * 2 + 1] = __builtin_amdgcn_wmma_f32_16x16x32_bf16(
            false, a1, false, b, (short)0, ha[j * 2 + 1], false, false);
      }
    }
#pragma unroll
    for (int j = 0; j < 3; ++j) {
      const int col = (wave * 3 + j) * 16 + mloc;     // 0..383 within chunk
#pragma unroll
      for (int s = 0; s < 2; ++s) {
#pragma unroll
        for (int v = 0; v < 8; ++v) {
          float hv = ha[j * 2 + s][v];
          hv = hv / (1.f + __expf(-hv));              // silu
          sH[(s * 16 + mrow + v) * HS + col] = f2bf(hv);
        }
      }
    }
    __syncthreads();

    // ---- phase B: acc += h[:, chunk] @ W2[chunk, :] ----
    const unsigned short* b2p[6];
#pragma unroll
    for (int j = 0; j < 6; ++j) {
      const int nh = (wave * 6 + j) * 16 + mloc;      // H index (B row)
      b2p[j] = w2T + ((size_t)e * H_DIM + nh) * F_DIM + chunk * F_CHUNK + kb;
    }
    for (int ks = 0; ks < 12; ++ks) {          // K = F_CHUNK = 384
      const int k0 = ks * 32;
      v16bf a0 = load_frag(aH0 + k0 + kb);
      v16bf a1 = load_frag(aH1 + k0 + kb);
#pragma unroll
      for (int j = 0; j < 6; ++j) {
        __builtin_prefetch(b2p[j] + k0 + 128, 0, 3);
        v16bf b = load_frag(b2p[j] + k0);
        acc[j * 2 + 0] = __builtin_amdgcn_wmma_f32_16x16x32_bf16(
            false, a0, false, b, (short)0, acc[j * 2 + 0], false, false);
        acc[j * 2 + 1] = __builtin_amdgcn_wmma_f32_16x16x32_bf16(
            false, a1, false, b, (short)0, acc[j * 2 + 1], false, false);
      }
    }
    __syncthreads();
  }

  // ---- weighted store of per-assignment outputs ----
#pragma unroll
  for (int j = 0; j < 6; ++j) {
    const int ncol = (wave * 6 + j) * 16 + mloc;
#pragma unroll
    for (int s = 0; s < 2; ++s) {
#pragma unroll
      for (int v = 0; v < 8; ++v) {
        const int m = s * 16 + mrow + v;
        if (mt * MB + m < cnt) {
          const int a = sA[m];
          ys[(size_t)a * H_DIM + ncol] = acc[j * 2 + s][v] * sW[m];
        }
      }
    }
  }
}

// ---------------- combine: out[t] = ys[2t] + ys[2t+1] ----------------
__global__ void k_combine(const float* __restrict__ ys, float* __restrict__ out) {
  const size_t i = (size_t)blockIdx.x * blockDim.x + threadIdx.x;  // float4 idx
  const size_t t = i / (H_DIM / 4);
  const size_t r = i % (H_DIM / 4);
  const float4* y4 = (const float4*)ys;
  float4 a = y4[(2 * t) * (H_DIM / 4) + r];
  float4 b = y4[(2 * t + 1) * (H_DIM / 4) + r];
  float4 o; o.x = a.x + b.x; o.y = a.y + b.y; o.z = a.z + b.z; o.w = a.w + b.w;
  ((float4*)out)[i] = o;
}

// ---------------- host launch ----------------
static constexpr size_t align256(size_t v) { return (v + 255) & ~(size_t)255; }

extern "C" void kernel_launch(void* const* d_in, const int* in_sizes, int n_in,
                              void* d_out, int out_size, void* d_ws, size_t ws_size,
                              hipStream_t stream) {
  const float* x  = (const float*)d_in[0];   // [T,H]
  const float* rw = (const float*)d_in[1];   // [H,E]
  const float* w1 = (const float*)d_in[2];   // [E,H,F]
  const float* w2 = (const float*)d_in[3];   // [E,F,H]
  float* out = (float*)d_out;                // [T,H]

  char* ws = (char*)d_ws;
  size_t off = 0;
  unsigned short* xb  = (unsigned short*)(ws + off); off = align256(off + (size_t)T_TOK * H_DIM * 2);
  unsigned short* w1T = (unsigned short*)(ws + off); off = align256(off + (size_t)E_EXP * F_DIM * H_DIM * 2);
  unsigned short* w2T = (unsigned short*)(ws + off); off = align256(off + (size_t)E_EXP * H_DIM * F_DIM * 2);
  float* assign_w = (float*)(ws + off);              off = align256(off + (size_t)TK * 4);
  int*   counts   = (int*)(ws + off);                off = align256(off + 64);
  int*   bucket   = (int*)(ws + off);                off = align256(off + (size_t)E_EXP * TK * 4);
  float* ys       = (float*)(ws + off);              off = align256(off + (size_t)TK * H_DIM * 4);
  (void)ws_size; (void)in_sizes; (void)n_in; (void)out_size;

  hipLaunchKernelGGL(k_init, dim3(1), dim3(32), 0, stream, counts);
  hipLaunchKernelGGL(k_router, dim3(T_TOK / 8), dim3(256), 0, stream,
                     x, rw, assign_w, counts, bucket);
  hipLaunchKernelGGL(k_convert_x, dim3((T_TOK * H_DIM / 4) / 256), dim3(256), 0, stream,
                     x, xb);
  hipLaunchKernelGGL(k_transpose_bf16,
                     dim3(E_EXP * (H_DIM / 32) * (F_DIM / 32)), dim3(256), 0, stream,
                     w1, w1T, H_DIM, F_DIM);
  hipLaunchKernelGGL(k_transpose_bf16,
                     dim3(E_EXP * (F_DIM / 32) * (H_DIM / 32)), dim3(256), 0, stream,
                     w2, w2T, F_DIM, H_DIM);
  hipLaunchKernelGGL(k_moe_mlp, dim3(E_EXP * MT_PER_E), dim3(256), 0, stream,
                     xb, w1T, w2T, assign_w, counts, bucket, ys);
  hipLaunchKernelGGL(k_combine, dim3((T_TOK * H_DIM / 4) / 256), dim3(256), 0, stream,
                     ys, out);
}
